// Ernie4_5_VL_MoeMoeBlock_10831907520597
// MI455X (gfx1250) — matrix operand order repaired and checked by reading, not compiled, not run on hardware
//
#include <hip/hip_runtime.h>
#include <hip/hip_bf16.h>

#define HID      2560
#define NEXP     16
#define TOPK     6
#define TEXT_I   1536
#define VIS_I    512
#define SHARED_I 3072

typedef __attribute__((ext_vector_type(16))) __bf16       v16bf;
typedef __attribute__((ext_vector_type(8)))  float        v8f;
typedef __attribute__((ext_vector_type(4)))  unsigned int u32x4;

static __device__ __forceinline__ __bf16 f2bf(float f) {
  return static_cast<__bf16>(f);   // native v_cvt_pk_bf16_f32 on gfx1250
}

// A-matrix (16x32 bf16, M x K) per ISA layout: lane<16 holds M=lane, K={0..7,16..23};
// lane>=16 holds M=lane-16, K={8..15,24..31}.  Two contiguous 16B chunks per lane.
static __device__ __forceinline__ v16bf loadA(const __bf16* rowbase, int k0, int lane) {
  const __bf16* p = rowbase + k0 + ((lane >> 4) << 3);
  union { v16bf v; u32x4 q[2]; } u;
  u.q[0] = *(const u32x4*)(p);
  u.q[1] = *(const u32x4*)(p + 16);
  return u.v;
}

// B-matrix (32x16, K x N) from row-major fp32 weights [K_total x ld]:
// lane holds column N = n0 + (lane&15); lanes 0-15 -> K {0..7,16..23}, lanes 16-31 -> K {8..15,24..31}.
static __device__ __forceinline__ v16bf loadBf32(const float* W, int ld, int k0, int n0, int lane) {
  const float* p = W + (size_t)(k0 + ((lane >> 4) << 3)) * ld + (n0 + (lane & 15));
  v16bf b;
#pragma unroll
  for (int j = 0; j < 16; ++j) b[j] = f2bf(p[(size_t)(j + ((j >> 3) << 3)) * ld]);
  return b;
}

static __device__ __forceinline__ v8f wmma_bf16(v16bf a, v16bf b, v8f c) {
  return __builtin_amdgcn_wmma_f32_16x16x32_bf16(false, a, false, b, (short)0, c, false, false);
}

static __device__ __forceinline__ float silu(float g) {
  return g * __builtin_amdgcn_rcpf(1.f + __expf(-g));
}

// Async DMA one 16B chunk: global -> LDS (ASYNCcnt-tracked, GV addressing).
static __device__ __forceinline__ void async_g2l_b128(unsigned lds_off, const void* gaddr) {
  asm volatile("global_load_async_to_lds_b128 %0, %1, off" :: "v"(lds_off), "v"(gaddr) : "memory");
}
static __device__ __forceinline__ void wait_async0() {
  asm volatile("s_wait_asynccnt 0x0" ::: "memory");
}
// Low 32 bits of a generic LDS pointer are the LDS byte address (aperture mapping).
static __device__ __forceinline__ unsigned lds_addr(const void* p) {
  return (unsigned)(unsigned long long)p;
}

// ---------------------------------------------------------------------------
__global__ void zero_kernel(int* p, int n) {
  int i = blockIdx.x * blockDim.x + threadIdx.x;
  if (i < n) p[i] = 0;
}

__global__ void tobf16_kernel(const float* __restrict__ in, __bf16* __restrict__ out, long long n) {
  long long i = (long long)blockIdx.x * blockDim.x + threadIdx.x;
  long long stride = (long long)gridDim.x * blockDim.x;
  for (; i < n; i += stride) out[i] = f2bf(in[i]);
}

// One wave per token: fp32 router, softmax, bias-corrected top-6, counts per bin.
__global__ void router_kernel(const float* __restrict__ hidden, const int* __restrict__ ttype,
                              const float* __restrict__ trw, const float* __restrict__ trb,
                              const float* __restrict__ vrw, const float* __restrict__ vrb,
                              float* __restrict__ out_logits, int* __restrict__ top_idx,
                              float* __restrict__ top_w, int* __restrict__ counts, int T) {
  int wave = (blockIdx.x * blockDim.x + threadIdx.x) >> 5;
  int lane = threadIdx.x & 31;
  if (wave >= T) return;
  int t = wave;
  int e = lane & 15, hh = lane >> 4;
  int m = ttype[t];
  const float* rw   = (m ? vrw : trw) + (size_t)e * HID;
  const float* bias = (m ? vrb : trb);
  const float* x    = hidden + (size_t)t * HID;
  float acc = 0.f;
  for (int h = hh; h < HID; h += 2) acc += x[h] * rw[h];
  acc += __shfl_xor(acc, 16, 32);
  float logit = acc;
  if (lane < NEXP) out_logits[(size_t)t * NEXP + lane] = logit;
  float mx = logit;
  for (int s = 8; s >= 1; s >>= 1) mx = fmaxf(mx, __shfl_xor(mx, s, 32));
  float ex = __expf(logit - mx);
  float sm = ex;
  for (int s = 8; s >= 1; s >>= 1) sm += __shfl_xor(sm, s, 32);
  float prob = ex / sm;
  float corrected = prob + bias[e];
  int   sel_i[TOPK];
  float sel_p[TOPK];
  float wsum = 0.f;
  for (int k = 0; k < TOPK; ++k) {
    float v = corrected; int idx = e;
    for (int s = 8; s >= 1; s >>= 1) {
      float ov = __shfl_xor(v, s, 32);
      int   oi = __shfl_xor(idx, s, 32);
      if (ov > v || (ov == v && oi < idx)) { v = ov; idx = oi; }
    }
    sel_i[k] = idx;
    sel_p[k] = __shfl(prob, idx, 32);
    wsum += sel_p[k];
    if (e == idx) corrected = -3.4e38f;
  }
  wsum = fmaxf(wsum, 1e-12f);
#pragma unroll
  for (int k = 0; k < TOPK; ++k) {
    if (lane == k) {
      top_idx[(size_t)t * TOPK + k] = sel_i[k];
      top_w[(size_t)t * TOPK + k]   = sel_p[k] / wsum;
    }
    if (lane == 0) atomicAdd(&counts[m * NEXP + sel_i[k]], 1);
  }
}

__global__ void scan_kernel(const int* __restrict__ counts, int* __restrict__ offs) {
  if (threadIdx.x == 0) {
    int o = 0;
    for (int b = 0; b < 32; ++b) { offs[b] = o; o += (counts[b] + 15) & ~15; }
    offs[32] = o;
  }
}

__global__ void scatter_kernel(const int* __restrict__ top_idx, const float* __restrict__ top_w,
                               const int* __restrict__ ttype, const int* __restrict__ offs,
                               int* __restrict__ cursor, int* __restrict__ slot_tok,
                               float* __restrict__ slot_w, int T) {
  int i = blockIdx.x * blockDim.x + threadIdx.x;
  if (i >= T * TOPK) return;
  int t = i / TOPK;
  int bin = ttype[t] * NEXP + top_idx[i];
  int pos = offs[bin] + atomicAdd(&cursor[bin], 1);
  slot_tok[pos] = t;
  slot_w[pos]   = top_w[i];
}

// Dynamic LDS: xs[16*HID] (async-staged A tile) + act[16*1536] = 128 KB.
#define LDS_BYTES ((16 * HID + 16 * TEXT_I) * 2)

// Shared expert: [16,H] x [H,3072] gate/up -> SiLU*u (bf16 LDS) -> x [3072,H].
__global__ __launch_bounds__(256)
void shared_expert_kernel(const __bf16* __restrict__ tok,
                          const float* __restrict__ gateW, const float* __restrict__ upW,
                          const float* __restrict__ downW, float* __restrict__ out, int T) {
  extern __shared__ __bf16 smem[];
  __bf16* xs  = smem;              // 16 x HID
  __bf16* act = smem + 16 * HID;   // 16 x 1536
  int t0 = blockIdx.x * 16;
  if (t0 >= T) return;
  int lane = threadIdx.x & 31, wv = threadIdx.x >> 5;
  int arow = lane & 15;

  // Async-stage the 16-row token tile into LDS (each wave: 2 rows, 10 x b128/lane).
  for (int r = wv; r < 16; r += 8) {
    int t = t0 + r; if (t >= T) t = T - 1;
    const char* src = (const char*)(tok + (size_t)t * HID);
    unsigned    dst = lds_addr(xs + r * HID);
    for (int i = lane; i < (HID * 2) / 16; i += 32)
      async_g2l_b128(dst + i * 16, src + i * 16);
  }
  wait_async0();
  __syncthreads();

  const __bf16* Arow = xs + (size_t)arow * HID;
  for (int hh = 0; hh < 2; ++hh) {
    int ibase = hh * 1536;
    for (int nt = wv; nt < 96; nt += 8) {
      v8f cg = {}; v8f cu = {};
#pragma unroll 2
      for (int k0 = 0; k0 < HID; k0 += 32) {
        v16bf a  = loadA(Arow, k0, lane);
        v16bf bg = loadBf32(gateW, SHARED_I, k0, ibase + nt * 16, lane);
        v16bf bu = loadBf32(upW,   SHARED_I, k0, ibase + nt * 16, lane);
        cg = wmma_bf16(a, bg, cg);
        cu = wmma_bf16(a, bu, cu);
      }
      int mrow = (lane >> 4) << 3, ncol = nt * 16 + (lane & 15);
#pragma unroll
      for (int v = 0; v < 8; ++v)
        act[(mrow + v) * 1536 + ncol] = f2bf(silu(cg[v]) * cu[v]);
    }
    __syncthreads();
    for (int nh = wv; nh < (HID >> 4); nh += 8) {
      v8f c = {};
#pragma unroll 2
      for (int k0 = 0; k0 < 1536; k0 += 32) {
        v16bf a  = loadA(&act[arow * 1536], k0, lane);
        v16bf bw = loadBf32(downW, HID, ibase + k0, nh * 16, lane);
        c = wmma_bf16(a, bw, c);
      }
      int mbase = (lane >> 4) << 3, ncol = nh * 16 + (lane & 15);
#pragma unroll
      for (int v = 0; v < 8; ++v) {
        int tt = t0 + mbase + v;
        if (tt < T) {
          float* dst = &out[(size_t)tt * HID + ncol];
          if (hh == 0) *dst = c[v];
          else         *dst += c[v];
        }
      }
    }
    __syncthreads();
  }
}

// Routed experts: one block = one 16-row tile of one (branch,expert) bin.
__global__ __launch_bounds__(256)
void moe_expert_kernel(const __bf16* __restrict__ tok,
                       const int* __restrict__ slot_tok, const float* __restrict__ slot_w,
                       const int* __restrict__ offs, const int* __restrict__ counts,
                       const float* __restrict__ t_gu, const float* __restrict__ t_dn,
                       const float* __restrict__ v_gu, const float* __restrict__ v_dn,
                       float* __restrict__ out) {
  extern __shared__ __bf16 smem[];
  __bf16* xs  = smem;              // 16 x HID
  __bf16* act = smem + 16 * HID;   // 16 x Idim (<= 1536)
  __shared__ int   trow[16];
  __shared__ float wrow[16];
  __shared__ int   sbin;
  int row0 = blockIdx.x * 16;
  if (threadIdx.x == 0) {
    int b = 0;
    while (b < 32 && row0 >= offs[b + 1]) ++b;
    sbin = (b < 32) ? b : -1;
  }
  __syncthreads();
  int b = sbin;
  if (b < 0) return;
  int cnt = counts[b], base = offs[b];
  if (threadIdx.x < 16) {
    int r = row0 + threadIdx.x;
    bool valid = (r - base) < cnt;
    trow[threadIdx.x] = valid ? slot_tok[r] : -1;
    wrow[threadIdx.x] = valid ? slot_w[r] : 0.f;
  }
  __syncthreads();

  int branch = b >> 4, e = b & 15;
  int Idim = branch ? VIS_I : TEXT_I;
  const float* gu = (branch ? v_gu : t_gu) + (size_t)e * HID * (size_t)(2 * Idim);
  const float* dn = (branch ? v_dn : t_dn) + (size_t)e * Idim * (size_t)HID;

  int lane = threadIdx.x & 31, wv = threadIdx.x >> 5;
  int arow = lane & 15;

  // Async-stage the gathered 16-row token tile into LDS.
  for (int r = wv; r < 16; r += 8) {
    int t = trow[r]; if (t < 0) t = 0;           // pad rows: stage row 0, output masked later
    const char* src = (const char*)(tok + (size_t)t * HID);
    unsigned    dst = lds_addr(xs + r * HID);
    for (int i = lane; i < (HID * 2) / 16; i += 32)
      async_g2l_b128(dst + i * 16, src + i * 16);
  }
  wait_async0();
  __syncthreads();

  const __bf16* Arow = xs + (size_t)arow * HID;

  // Phase 1: gate/up GEMM + SiLU*u -> act (bf16 in LDS)
  int NI = Idim >> 4;
  for (int nt = wv; nt < NI; nt += 8) {
    v8f cg = {}; v8f cu = {};
#pragma unroll 2
    for (int k0 = 0; k0 < HID; k0 += 32) {
      v16bf a  = loadA(Arow, k0, lane);
      v16bf bg = loadBf32(gu, 2 * Idim, k0, nt * 16, lane);
      v16bf bu = loadBf32(gu, 2 * Idim, k0, Idim + nt * 16, lane);
      cg = wmma_bf16(a, bg, cg);
      cu = wmma_bf16(a, bu, cu);
    }
    int mrow = (lane >> 4) << 3, ncol = nt * 16 + (lane & 15);
#pragma unroll
    for (int v = 0; v < 8; ++v)
      act[(mrow + v) * Idim + ncol] = f2bf(silu(cg[v]) * cu[v]);
  }
  __syncthreads();

  // Phase 2: down GEMM, scale by routing weight, accumulate into final output.
  for (int nh = wv; nh < (HID >> 4); nh += 8) {
    v8f c = {};
#pragma unroll 2
    for (int k0 = 0; k0 < Idim; k0 += 32) {
      v16bf a  = loadA(&act[arow * Idim], k0, lane);
      v16bf bw = loadBf32(dn, HID, k0, nh * 16, lane);
      c = wmma_bf16(a, bw, c);
    }
    int mbase = (lane >> 4) << 3, ncol = nh * 16 + (lane & 15);
#pragma unroll
    for (int v = 0; v < 8; ++v) {
      int mm = mbase + v;
      int tt = trow[mm];
      if (tt >= 0)
        unsafeAtomicAdd(&out[(size_t)tt * HID + ncol], c[v] * wrow[mm]);
    }
  }
}

// ---------------------------------------------------------------------------
extern "C" void kernel_launch(void* const* d_in, const int* in_sizes, int n_in,
                              void* d_out, int out_size, void* d_ws, size_t ws_size,
                              hipStream_t stream) {
  const float* hidden = (const float*)d_in[0];
  const int*   ttype  = (const int*)d_in[1];
  const float* trw    = (const float*)d_in[2];
  const float* trb    = (const float*)d_in[3];
  const float* tgu    = (const float*)d_in[4];
  const float* tdn    = (const float*)d_in[5];
  const float* vrw    = (const float*)d_in[6];
  const float* vrb    = (const float*)d_in[7];
  const float* vgu    = (const float*)d_in[8];
  const float* vdn    = (const float*)d_in[9];
  const float* sg     = (const float*)d_in[10];
  const float* su     = (const float*)d_in[11];
  const float* sd     = (const float*)d_in[12];
  (void)n_in; (void)out_size; (void)ws_size;

  const int T = in_sizes[0] / HID;
  float* out_final  = (float*)d_out;
  float* out_logits = out_final + (size_t)T * HID;

  char* ws = (char*)d_ws;
  size_t off = 0;
  auto carve = [&](size_t bytes) -> void* {
    void* p = ws + off;
    off = (off + bytes + 255) & ~(size_t)255;
    return p;
  };
  __bf16* tok      = (__bf16*)carve((size_t)T * HID * 2);
  int*    top_idx  = (int*)  carve((size_t)T * TOPK * 4);
  float*  top_w    = (float*)carve((size_t)T * TOPK * 4);
  int*    slot_tok = (int*)  carve(((size_t)T * TOPK + 512) * 4);
  float*  slot_w   = (float*)carve(((size_t)T * TOPK + 512) * 4);
  int*    binmeta  = (int*)  carve(128 * 4);
  int* counts = binmeta;        // 32
  int* cursor = binmeta + 32;   // 32
  int* offs   = binmeta + 64;   // 33

  zero_kernel<<<1, 64, 0, stream>>>(binmeta, 64);
  tobf16_kernel<<<4096, 256, 0, stream>>>(hidden, tok, (long long)T * HID);
  router_kernel<<<(T + 7) / 8, 256, 0, stream>>>(hidden, ttype, trw, trb, vrw, vrb,
                                                 out_logits, top_idx, top_w, counts, T);
  scan_kernel<<<1, 32, 0, stream>>>(counts, offs);
  scatter_kernel<<<(T * TOPK + 255) / 256, 256, 0, stream>>>(top_idx, top_w, ttype, offs,
                                                             cursor, slot_tok, slot_w, T);
  shared_expert_kernel<<<(T + 15) / 16, 256, LDS_BYTES, stream>>>(tok, sg, su, sd, out_final, T);
  int maxtiles = (T * TOPK + 32 * 15 + 15) / 16;
  moe_expert_kernel<<<maxtiles, 256, LDS_BYTES, stream>>>(tok, slot_tok, slot_w, offs, counts,
                                                          tgu, tdn, vgu, vdn, out_final);
}
